// GNNChildEncoder_26577257628365
// MI455X (gfx1250) — matrix operand order repaired and checked by reading, not compiled, not run on hardware
//
#include <hip/hip_runtime.h>
#include <hip/hip_bf16.h>
#include <stdint.h>

#define N_CHILDS   50000
#define N_EDGES    500000
#define NODE_FEAT  256
#define NUM_SEM    57
#define HID        256
#define EDGE_FEAT  64
#define EDGE_TYPES 4
#define D_IN       (NODE_FEAT + NUM_SEM)                 /* 313 */
#define D_NE       (2*HID + EDGE_FEAT + EDGE_TYPES)      /* 580 */
#define KP_CHILD   320                                   /* 313 padded to 10*32 */
#define KP_EDGE    608                                   /* 580 padded to 19*32 */
#define NT_TOTAL   (HID/16)                              /* 16 N-tiles of 16 */
#define EM         64                                    /* edges per workgroup */

typedef __attribute__((ext_vector_type(16))) __bf16 v16bf;
typedef __attribute__((ext_vector_type(8)))  __bf16 v8bf;
typedef __attribute__((ext_vector_type(8)))  float  v8f;
typedef __attribute__((ext_vector_type(4)))  int    v4i;

// CDNA5 async global->LDS path (ASYNCcnt), with safe fallback if the builtin
// is not declared by this toolchain. Parameter types per hipcc diagnostic:
// (AS1 int4*, AS3 int4*, imm int offset, imm int cpol).
#if defined(__HIP_DEVICE_COMPILE__) && \
    __has_builtin(__builtin_amdgcn_global_load_async_to_lds_b128)
#define HAVE_ASYNC_LDS 1
#define GPTR(p) ((__attribute__((address_space(1))) v4i*)(p))
#define LPTR(p) ((__attribute__((address_space(3))) v4i*)(p))
#endif

__device__ __forceinline__ unsigned short f2bf(float f) {
  unsigned u = __float_as_uint(f);
  u += 0x7FFFu + ((u >> 16) & 1u);        // round-to-nearest-even
  return (unsigned short)(u >> 16);
}

// ---------------- generic zero ----------------
__global__ void zero_k(float* p, long long n) {
  long long i = (long long)blockIdx.x * blockDim.x + threadIdx.x;
  long long stride = (long long)gridDim.x * blockDim.x;
  for (; i < n; i += stride) p[i] = 0.0f;
}

// ---------------- weight pre-pack to bf16 WMMA B-fragment layout ----------------
// out layout: [KP/32][NT=16][lane=32][elem=16] bf16; lane<16 covers K=k0..k0+15,
// lanes 16..31 cover K=k0+16..k0+31 of column n0+(lane&15). Zero-pad K >= Kin.
__global__ void pack_w(const float* __restrict__ W, int Kin, int KP,
                       unsigned short* __restrict__ out) {
  int idx = blockIdx.x * blockDim.x + threadIdx.x;
  int total = KP * HID;
  if (idx >= total) return;
  int e    =  idx        & 15;
  int lane = (idx >> 4)  & 31;
  int nt   = (idx >> 9)  & 15;
  int kb   =  idx >> 13;
  int n = nt * 16 + (lane & 15);
  int k = kb * 32 + ((lane >> 4) << 4) + e;
  float v = (k < Kin) ? W[(size_t)k * HID + n] : 0.0f;
  out[idx] = f2bf(v);
}

// ---------------- degree (segment counts over src) ----------------
__global__ void degree_k(const long long* __restrict__ eidx, float* __restrict__ deg, int nE) {
  int i = blockIdx.x * blockDim.x + threadIdx.x;
  if (i < nE) atomicAdd(&deg[(int)eidx[2LL * i]], 1.0f);
}

// ---------------- child encoder: x = relu(cf @ Wc + b) * exists, WMMA bf16 ----------------
__global__ __launch_bounds__(128)
void child_enc_wmma(const float* __restrict__ cf, const float* __restrict__ cex,
                    const unsigned short* __restrict__ Wp, const float* __restrict__ bias,
                    unsigned short* __restrict__ xbf, unsigned int* __restrict__ parentU) {
  __shared__ __align__(16) unsigned short Atile[16][KP_CHILD];
  __shared__ float s_ce[16];
  const int t = threadIdx.x;
  const long long tile = blockIdx.x;
  const int row = t >> 3, l8 = t & 7;
  if (t < 16) s_ce[t] = cex[tile * 16 + t];
  {
    long long n = tile * 16 + row;
    const float* src = cf + n * D_IN;
    for (int jj = 0; jj < KP_CHILD / 8; ++jj) {
      int j = l8 * (KP_CHILD / 8) + jj;
      Atile[row][j] = f2bf(j < D_IN ? src[j] : 0.0f);
    }
  }
  __syncthreads();

  const int lane = t & 31, wv = t >> 5;
  const int colInTile = lane & 15;
  const int halfsel = lane >> 4;
  v8f acc[4];
  for (int i = 0; i < 4; ++i) {
    float b = bias[(wv * 4 + i) * 16 + colInTile];
    for (int r = 0; r < 8; ++r) acc[i][r] = b;
  }
  for (int kb = 0; kb < KP_CHILD / 32; ++kb) {
    union { v16bf v; v8bf h[2]; } a;
    int k0 = kb * 32 + halfsel * 16;
    a.h[0] = *(const v8bf*)&Atile[colInTile][k0];
    a.h[1] = *(const v8bf*)&Atile[colInTile][k0 + 8];
    if (kb + 1 < KP_CHILD / 32)
      __builtin_prefetch(Wp + (size_t)((kb + 1) * NT_TOTAL + wv * 4) * 32 * 16, 0, 1);
    for (int i = 0; i < 4; ++i) {
      union { v16bf v; uint4 u[2]; } b;
      const uint4* bp = (const uint4*)Wp +
          ((size_t)(kb * NT_TOTAL + (wv * 4 + i)) * 32 + lane) * 2;
      b.u[0] = bp[0]; b.u[1] = bp[1];
      acc[i] = __builtin_amdgcn_wmma_f32_16x16x32_bf16(
          false, a.v, false, b.v, (short)0, acc[i], false, false);
    }
  }
  // epilogue: relu, * exists, store bf16 x, column-max into parent slot 0
  for (int i = 0; i < 4; ++i) {
    int col = (wv * 4 + i) * 16 + colInTile;
    float vmax = 0.0f;
    for (int r = 0; r < 8; ++r) {
      int m = halfsel * 8 + r;
      float v = fmaxf(acc[i][r], 0.0f) * s_ce[m];
      xbf[(tile * 16 + m) * (size_t)HID + col] = f2bf(v);
      vmax = fmaxf(vmax, v);
    }
    atomicMax(&parentU[col], __float_as_uint(vmax));  // all values >= 0
  }
}

// ---------------- edge MLP: relu([x[s],x[d],ef] @ W + b) scatter-added to accum[s] ---
// 64 edges x 256 outputs per workgroup; 4 waves, each owning 64 columns.
// B-fragments are reused across the 4 M-subtiles (4 WMMAs per loaded fragment).
__global__ __launch_bounds__(128)
void edge_mlp_wmma(const unsigned short* __restrict__ xbf,
                   const float* __restrict__ eto, const float* __restrict__ efeat,
                   const long long* __restrict__ eidx,
                   const unsigned short* __restrict__ Wp, const float* __restrict__ bias,
                   float* __restrict__ accum) {
  __shared__ __align__(16) unsigned short Atile[EM][KP_EDGE];
  __shared__ int s_src[EM];
  __shared__ int s_dst[EM];
  const int t = threadIdx.x;
  const long long tile = blockIdx.x;
  const long long ebase = tile * EM;
  if (t < EM) {
    long long e = ebase + t;
    if (e < N_EDGES) {
      s_src[t] = (int)eidx[2 * e];
      s_dst[t] = (int)eidx[2 * e + 1];
    } else {
      s_src[t] = 0;                    // tail tile: clamp to node 0, scatter guarded
      s_dst[t] = 0;
    }
  }
  __syncthreads();
  const int row = t >> 1, half = t & 1;    // 2 threads per edge row
  { // gather x[src] -> cols [0,256), x[dst] -> [256,512): 256B per thread each
    const uint4* ps = (const uint4*)(xbf + (size_t)s_src[row] * HID) + half * 16;
    uint4*       qs = (uint4*)&Atile[row][half * 128];
    const uint4* pd = (const uint4*)(xbf + (size_t)s_dst[row] * HID) + half * 16;
    uint4*       qd = (uint4*)&Atile[row][HID + half * 128];
#if HAVE_ASYNC_LDS
    for (int i = 0; i < 16; ++i) {
      __builtin_amdgcn_global_load_async_to_lds_b128(GPTR(ps + i), LPTR(qs + i), 0, 0);
      __builtin_amdgcn_global_load_async_to_lds_b128(GPTR(pd + i), LPTR(qd + i), 0, 0);
    }
#else
    for (int i = 0; i < 16; ++i) { qs[i] = ps[i]; qd[i] = pd[i]; }
#endif
  }
  { // edge features -> cols [512,580), zero pad -> 608 (48 cols per thread)
    long long e = ebase + row;
    const float* pe = eto + e * EDGE_TYPES;
    const float* pf = efeat + e * EDGE_FEAT;
    for (int jj = 0; jj < 48; ++jj) {
      int j = half * 48 + jj;
      float v = 0.0f;
      if (j < EDGE_TYPES)                  v = pe[j];
      else if (j < EDGE_TYPES + EDGE_FEAT) v = pf[j - EDGE_TYPES];
      Atile[row][2 * HID + j] = f2bf(v);
    }
  }
#if HAVE_ASYNC_LDS
#if __has_builtin(__builtin_amdgcn_s_wait_asynccnt)
  __builtin_amdgcn_s_wait_asynccnt(0);
#else
  asm volatile("s_wait_asynccnt 0" ::: "memory");
#endif
#endif
  __syncthreads();

  const int lane = t & 31, wv = t >> 5;
  const int colInTile = lane & 15;
  const int halfsel = lane >> 4;
  v8f acc[4][4];                            // [m-subtile][n-tile]
  for (int i = 0; i < 4; ++i) {
    float b = bias[(wv * 4 + i) * 16 + colInTile];
    for (int mt = 0; mt < 4; ++mt)
      for (int r = 0; r < 8; ++r) acc[mt][i][r] = b;
  }
  for (int kb = 0; kb < KP_EDGE / 32; ++kb) {
    union { v16bf v; uint4 u[2]; } b[4];
    for (int i = 0; i < 4; ++i) {
      const uint4* bp = (const uint4*)Wp +
          ((size_t)(kb * NT_TOTAL + (wv * 4 + i)) * 32 + lane) * 2;
      b[i].u[0] = bp[0]; b[i].u[1] = bp[1];
    }
    if (kb + 1 < KP_EDGE / 32)
      __builtin_prefetch(Wp + (size_t)((kb + 1) * NT_TOTAL + wv * 4) * 32 * 16, 0, 1);
    const int k0 = kb * 32 + halfsel * 16;
    for (int mt = 0; mt < 4; ++mt) {
      union { v16bf v; v8bf h[2]; } a;
      const int am = mt * 16 + colInTile;
      a.h[0] = *(const v8bf*)&Atile[am][k0];
      a.h[1] = *(const v8bf*)&Atile[am][k0 + 8];
      for (int i = 0; i < 4; ++i)
        acc[mt][i] = __builtin_amdgcn_wmma_f32_16x16x32_bf16(
            false, a.v, false, b[i].v, (short)0, acc[mt][i], false, false);
    }
  }
  // relu + scatter-add into accum[src] (tail tile rows guarded)
  for (int mt = 0; mt < 4; ++mt) {
    for (int i = 0; i < 4; ++i) {
      int col = (wv * 4 + i) * 16 + colInTile;
      for (int r = 0; r < 8; ++r) {
        int m = mt * 16 + halfsel * 8 + r;
        if (ebase + m < N_EDGES) {
          float v = fmaxf(acc[mt][i][r], 0.0f);
          atomicAdd(&accum[(size_t)s_src[m] * HID + col], v);
        }
      }
    }
  }
}

// ---------------- segment-mean + column-max + re-zero accumulator ----------------
__global__ __launch_bounds__(256)
void finalize_k(float* __restrict__ accum, const float* __restrict__ deg,
                unsigned short* __restrict__ xbf, unsigned int* __restrict__ parentU,
                int itSlot) {
  int c = threadIdx.x;           // column 0..255 (coalesced)
  int r0 = blockIdx.x * 256;
  float vmax = 0.0f;
  for (int rr = 0; rr < 256; ++rr) {
    int n = r0 + rr;
    if (n >= N_CHILDS) break;
    size_t idx = (size_t)n * HID + c;
    float v = accum[idx] / fmaxf(deg[n], 1.0f);   // sums of relu -> v >= 0
    accum[idx] = 0.0f;                            // ready for next iteration
    xbf[idx] = f2bf(v);
    vmax = fmaxf(vmax, v);
  }
  atomicMax(&parentU[itSlot * HID + c], __float_as_uint(vmax));
}

// ---------------- final: out = relu(parent(1x768) @ Wp + bp) ----------------
__global__ __launch_bounds__(256)
void final_k(const unsigned int* __restrict__ parentU, const float* __restrict__ Wp,
             const float* __restrict__ bp, float* __restrict__ out) {
  __shared__ float sp[3 * HID];
  for (int i = threadIdx.x; i < 3 * HID; i += blockDim.x)
    sp[i] = __uint_as_float(parentU[i]);          // values >= 0, bits == float
  __syncthreads();
  int j = threadIdx.x;
  float s = bp[j];
  for (int i = 0; i < 3 * HID; ++i) s += sp[i] * Wp[(size_t)i * HID + j];
  out[j] = fmaxf(s, 0.0f);
}

extern "C" void kernel_launch(void* const* d_in, const int* in_sizes, int n_in,
                              void* d_out, int out_size, void* d_ws, size_t ws_size,
                              hipStream_t stream) {
  const float*     child_feats = (const float*)d_in[0];
  const float*     child_ex    = (const float*)d_in[1];
  const float*     eto         = (const float*)d_in[2];
  const float*     efeat       = (const float*)d_in[3];
  const long long* eidx        = (const long long*)d_in[4];
  const float*     W_child     = (const float*)d_in[5];
  const float*     b_child     = (const float*)d_in[6];
  const float*     W_ne0       = (const float*)d_in[7];
  const float*     b_ne0       = (const float*)d_in[8];
  const float*     W_ne1       = (const float*)d_in[9];
  const float*     b_ne1       = (const float*)d_in[10];
  const float*     W_parent    = (const float*)d_in[11];
  const float*     b_parent    = (const float*)d_in[12];

  uint8_t* ws = (uint8_t*)d_ws;
  size_t off = 0;
  auto take = [&](size_t bytes) -> void* {
    void* p = ws + off;
    off += (bytes + 255) & ~(size_t)255;
    return p;
  };
  unsigned short* xbf     = (unsigned short*)take((size_t)N_CHILDS * HID * 2);
  float*          accum   = (float*)take((size_t)N_CHILDS * HID * 4);
  float*          deg     = (float*)take((size_t)N_CHILDS * 4);
  unsigned int*   parentU = (unsigned int*)take(3 * HID * 4);
  unsigned short* WcP     = (unsigned short*)take((size_t)KP_CHILD * HID * 2);
  unsigned short* W0P     = (unsigned short*)take((size_t)KP_EDGE * HID * 2);
  unsigned short* W1P     = (unsigned short*)take((size_t)KP_EDGE * HID * 2);
  (void)ws_size; (void)in_sizes; (void)n_in; (void)out_size;

  // 1) zero scratch we depend on (workspace is not re-poisoned between calls)
  zero_k<<<2048, 256, 0, stream>>>(accum, (long long)N_CHILDS * HID);
  zero_k<<<64,   256, 0, stream>>>(deg, N_CHILDS);
  zero_k<<<1,    256, 0, stream>>>((float*)parentU, 3 * HID);

  // 2) pack weights -> bf16 WMMA B-fragments (K zero-padded)
  pack_w<<<(KP_CHILD * HID + 255) / 256, 256, 0, stream>>>(W_child, D_IN, KP_CHILD, WcP);
  pack_w<<<(KP_EDGE  * HID + 255) / 256, 256, 0, stream>>>(W_ne0,   D_NE, KP_EDGE,  W0P);
  pack_w<<<(KP_EDGE  * HID + 255) / 256, 256, 0, stream>>>(W_ne1,   D_NE, KP_EDGE,  W1P);

  // 3) child encoder (WMMA) -> xbf, parent slot 0
  child_enc_wmma<<<N_CHILDS / 16, 128, 0, stream>>>(child_feats, child_ex, WcP, b_child,
                                                    xbf, parentU);
  // 4) degrees
  degree_k<<<(N_EDGES + 255) / 256, 256, 0, stream>>>(eidx, deg, N_EDGES);

  const int egrid = (N_EDGES + EM - 1) / EM;   // 7813 (last tile has 32 edges)
  // 5) iteration 1
  edge_mlp_wmma<<<egrid, 128, 0, stream>>>(xbf, eto, efeat, eidx, W0P, b_ne0, accum);
  finalize_k<<<(N_CHILDS + 255) / 256, 256, 0, stream>>>(accum, deg, xbf, parentU, 1);

  // 6) iteration 2
  edge_mlp_wmma<<<egrid, 128, 0, stream>>>(xbf, eto, efeat, eidx, W1P, b_ne1, accum);
  finalize_k<<<(N_CHILDS + 255) / 256, 256, 0, stream>>>(accum, deg, xbf, parentU, 2);

  // 7) output head
  final_k<<<1, 256, 0, stream>>>(parentU, W_parent, b_parent, (float*)d_out);
}
